// Model_51977694216628
// MI455X (gfx1250) — matrix operand order repaired and checked by reading, not compiled, not run on hardware
//
#include <hip/hip_runtime.h>

typedef _Float16 v16h __attribute__((ext_vector_type(16)));
typedef float    v8f  __attribute__((ext_vector_type(8)));

#define WMMA_F16(a, b, c) \
  __builtin_amdgcn_wmma_f32_16x16x32_f16(false, (a), false, (b), (short)0, (c), false, false)

namespace {
constexpr int kN    = 262144;   // nodes
constexpr int kG    = 256;      // graphs
constexpr int kP    = 1024;     // nodes per graph
constexpr int kE    = 4194304;  // edges
constexpr int kFin  = 128;
constexpr int kH    = 32;
constexpr int kC    = 10;
constexpr int kOut  = 16;
constexpr int kIn   = 96;       // 3*H
constexpr int kCol  = 160;      // C*OUT_LEN
constexpr int kCap  = 24576;    // per-graph edge bucket capacity (mean 16384)
constexpr int kIters = 3;
}

// ---------------- degree / bucketing ----------------

__global__ void k_init_deg(float* deg) {
  int i = blockIdx.x * blockDim.x + threadIdx.x;
  if (i < kN) deg[i] = 1.0f;   // GCN re-adds self loop with weight 1
}

__global__ void k_edge_deg(const int* __restrict__ src, const int* __restrict__ dst,
                           float* deg) {
  int e = blockIdx.x * blockDim.x + threadIdx.x;
  if (e >= kE) return;
  int s = src[e], d = dst[e];
  if (s != d) atomicAdd(&deg[d], 1.0f);
}

__global__ void k_deg_to_dis(float* deg) {
  int i = blockIdx.x * blockDim.x + threadIdx.x;
  if (i < kN) deg[i] = rsqrtf(deg[i]);   // in place: now holds deg^{-1/2}
}

__global__ void k_zero_counts(int* count) {
  int i = blockIdx.x * blockDim.x + threadIdx.x;
  if (i < kG) count[i] = 0;
}

__global__ void k_bucket_fill(const int* __restrict__ src, int* count, int* bucket) {
  int e = blockIdx.x * blockDim.x + threadIdx.x;
  if (e >= kE) return;
  int g = src[e] >> 10;                       // both endpoints in same graph
  int pos = atomicAdd(&count[g], 1);
  if (pos < kCap) bucket[(size_t)g * kCap + pos] = e;
}

// ---- helper: pack 8 contiguous f32 (2x b128 loads) into f16 fragment slots -------

__device__ inline void loadPack8(const float* p, v16h& a, int base) {
  float4 q0 = ((const float4*)p)[0];
  float4 q1 = ((const float4*)p)[1];
  a[base + 0] = (_Float16)q0.x; a[base + 1] = (_Float16)q0.y;
  a[base + 2] = (_Float16)q0.z; a[base + 3] = (_Float16)q0.w;
  a[base + 4] = (_Float16)q1.x; a[base + 5] = (_Float16)q1.y;
  a[base + 6] = (_Float16)q1.z; a[base + 7] = (_Float16)q1.w;
}

__device__ inline void loadPack8Add(const float* p, const float* q, v16h& a, int base) {
  float4 x0 = ((const float4*)p)[0];
  float4 x1 = ((const float4*)p)[1];
  float4 e0 = ((const float4*)q)[0];
  float4 e1 = ((const float4*)q)[1];
  a[base + 0] = (_Float16)(x0.x + e0.x); a[base + 1] = (_Float16)(x0.y + e0.y);
  a[base + 2] = (_Float16)(x0.z + e0.z); a[base + 3] = (_Float16)(x0.w + e0.w);
  a[base + 4] = (_Float16)(x1.x + e1.x); a[base + 5] = (_Float16)(x1.y + e1.y);
  a[base + 6] = (_Float16)(x1.z + e1.z); a[base + 7] = (_Float16)(x1.w + e1.w);
}

// ---------------- dense GEMM: Y[N,32] = X[N,K] @ W[K,32] (f16 WMMA, fp32 acc) --------

__global__ __launch_bounds__(256) void k_gemm_h(const float* __restrict__ X, int ldx,
                                                int K, const float* __restrict__ W,
                                                float* __restrict__ Y) {
  const int lane = threadIdx.x & 31;
  const int wv   = threadIdx.x >> 5;
  const int hi   = lane >> 4;        // half-wave
  const int m    = lane & 15;
  const int row0 = (blockIdx.x * 8 + wv) * 16;
  const float* xr = X + (size_t)(row0 + m) * ldx;
  v8f acc0 = {}, acc1 = {};
  for (int kb = 0; kb < K; kb += 32) {
    v16h a;
    const int ka = kb + hi * 16;     // A: lanes<16 K0-15, lanes>=16 K16-31 (matches B)
    loadPack8(xr + ka, a, 0);        // b128 loads (32B-aligned offsets)
    loadPack8(xr + ka + 8, a, 8);
    v16h b0, b1;
    const int kB = kb + hi * 16;     // B: lanes<16 K0-15, lanes>=16 K16-31, N=lane&15
#pragma unroll
    for (int i = 0; i < 16; ++i) {
      const float* wr = W + (size_t)(kB + i) * kH + m;
      b0[i] = (_Float16)wr[0];
      b1[i] = (_Float16)wr[16];
    }
    acc0 = WMMA_F16(a, b0, acc0);
    acc1 = WMMA_F16(a, b1, acc1);
  }
#pragma unroll
  for (int r = 0; r < 8; ++r) {
    const int mr = r + hi * 8;       // C/D layout: VGPR r -> M=r (+8 upper half)
    float* yr = Y + (size_t)(row0 + mr) * kH + m;
    yr[0]  = acc0[r];
    yr[16] = acc1[r];
  }
}

// ---------------- per-graph GCN aggregation in LDS + fused epilogue ----------------

__global__ __launch_bounds__(256) void k_gcn_agg(const float* __restrict__ hlin,
                                                 const float* __restrict__ dis,
                                                 const int* __restrict__ src,
                                                 const int* __restrict__ dst,
                                                 const int* __restrict__ bucket,
                                                 const int* __restrict__ count,
                                                 const float* __restrict__ bias,
                                                 float* __restrict__ Hcat, int colOff) {
  __shared__ float ag[kP * kH];      // 128 KB LDS accumulator for this graph
  const int g = blockIdx.x;
  const int t = threadIdx.x;
  const int lane = t & 31;           // channel
  const int wv = t >> 5;
  for (int i = t; i < kP * kH; i += 256) ag[i] = 0.0f;
  __syncthreads();
  const int ec = min(count[g], kCap);
  const int* bk = bucket + (size_t)g * kCap;
  for (int e = wv; e < ec; e += 8) {           // one wave per edge, lane = channel
    int eid = bk[e];
    int s = src[eid], d = dst[eid];
    if (s != d) {
      float w = dis[s] * dis[d];
      float v = hlin[(size_t)s * kH + lane] * w;   // L2-resident graph slice
      atomicAdd(&ag[(d & (kP - 1)) * kH + lane], v);
    }
  }
  __syncthreads();
  for (int i = t; i < kP * kH; i += 256) {
    int nl = i >> 5, ch = i & 31;
    int node = g * kP + nl;
    float di = dis[node];
    float v = ag[i] + hlin[(size_t)node * kH + ch] * (di * di) + bias[ch];
    Hcat[(size_t)node * kIn + colOff + ch] = tanhf(v);
  }
}

// ---------------- position encoding table ----------------

__global__ void k_posenc(float* pe) {
  int i = blockIdx.x * blockDim.x + threadIdx.x;
  if (i >= kP * kIn) return;
  int p = i / kIn, d = i % kIn;
  int pr = d >> 1;
  // ang = p / 10000^(2*pr/96)
  float ang = (float)p * __expf(-((float)(2 * pr) / (float)kIn) * 9.210340371976184f);
  pe[i] = (d & 1) ? __cosf(ang) : __sinf(ang);
}

// ---------------- per-graph bitonic sort-pool (descending by last channel) ----------

__global__ __launch_bounds__(512) void k_sort(const float* __restrict__ Hcat,
                                              int* __restrict__ order) {
  __shared__ float key[kP];
  __shared__ int   idx[kP];
  const int g = blockIdx.x;
  const int t = threadIdx.x;
  for (int i = t; i < kP; i += 512) {
    key[i] = -Hcat[(size_t)(g * kP + i) * kIn + (kIn - 1)];  // ascending on -key
    idx[i] = i;
  }
  for (int k2 = 2; k2 <= kP; k2 <<= 1) {
    for (int j = k2 >> 1; j > 0; j >>= 1) {
      __syncthreads();
      for (int i = t; i < kP; i += 512) {
        int l = i ^ j;
        if (l > i) {
          bool up = ((i & k2) == 0);
          float ki = key[i], kl = key[l];
          bool sw = up ? (ki > kl) : (ki < kl);
          if (sw) {
            key[i] = kl; key[l] = ki;
            int ti = idx[i]; idx[i] = idx[l]; idx[l] = ti;
          }
        }
      }
    }
  }
  __syncthreads();
  for (int i = t; i < kP; i += 512) order[g * kP + i] = idx[i];
}

// ------- priors GEMM: [N,96] (gathered+PE) @ capsW^T [96,160] -> priors [N,160] -----

__global__ __launch_bounds__(256) void k_gemm_priors(const float* __restrict__ Hcat,
                                                     const int* __restrict__ order,
                                                     const float* __restrict__ pe,
                                                     const float* __restrict__ capsW,
                                                     float* __restrict__ priors) {
  const int lane = threadIdx.x & 31;
  const int wv   = threadIdx.x >> 5;
  const int hi   = lane >> 4;
  const int m    = lane & 15;
  const int row0 = (blockIdx.x * 8 + wv) * 16;
  const int row  = row0 + m;                    // sorted-order global row
  const int g    = row >> 10;
  const int p    = row & (kP - 1);
  const int node = (g << 10) + order[row];      // gather through sort permutation
  const float* xr  = Hcat + (size_t)node * kIn;
  const float* per = pe + (size_t)p * kIn;
  v8f acc[10] = {};
  for (int kb = 0; kb < kIn; kb += 32) {
    v16h a;
    const int ka = kb + hi * 16;
    loadPack8Add(xr + ka,      per + ka,      a, 0);   // b128 loads + PE add
    loadPack8Add(xr + ka + 8,  per + ka + 8,  a, 8);
    const int kB = kb + hi * 16;
#pragma unroll
    for (int t2 = 0; t2 < 10; ++t2) {
      v16h b;
      const int n = t2 * 16 + m;                // column = o*16+j ; B[k][n] = capsW[n*96+k]
      const float* wp = capsW + (size_t)n * kIn + kB;   // 64B-aligned, 16 contiguous
      loadPack8(wp, b, 0);
      loadPack8(wp + 8, b, 8);
      acc[t2] = WMMA_F16(a, b, acc[t2]);
    }
  }
#pragma unroll
  for (int t2 = 0; t2 < 10; ++t2) {
#pragma unroll
    for (int r = 0; r < 8; ++r) {
      priors[(size_t)(row0 + r + hi * 8) * kCol + t2 * 16 + m] = acc[t2][r];
    }
  }
}

// ---------------- fused k-means routing + squash-norm, one block per (g, o) ---------

__device__ inline float waveSum(float v) {
#pragma unroll
  for (int s = 16; s > 0; s >>= 1) v += __shfl_xor(v, s, 32);
  return v;
}
__device__ inline float waveMax(float v) {
#pragma unroll
  for (int s = 16; s > 0; s >>= 1) v = fmaxf(v, __shfl_xor(v, s, 32));
  return v;
}

__global__ __launch_bounds__(256) void k_routing(const float* __restrict__ priors,
                                                 float* __restrict__ out) {
  __shared__ float pri[kP * kOut];   // 64 KB: the (g,o) prior slice
  __shared__ float vacc[kOut];
  __shared__ float wred[8];
  const int t = threadIdx.x;
  const int lane = t & 31;
  const int wv = t >> 5;
  const int g = blockIdx.x / kC;
  const int o = blockIdx.x % kC;
  const float* base = priors + (size_t)g * kP * kCol + o * kOut;

  for (int k = t; k < kP; k += 256) {          // 64B-aligned float4 loads
    const float4* s4 = (const float4*)(base + (size_t)k * kCol);
    float4* d4 = (float4*)&pri[k * kOut];
    d4[0] = s4[0]; d4[1] = s4[1]; d4[2] = s4[2]; d4[3] = s4[3];
  }
  if (t < kOut) vacc[t] = 0.0f;
  __syncthreads();

  const int k0 = t * 4;                        // 4 rows per thread
  float vloc[kOut];
  {                                            // v = mean_k priors
    float part[kOut];
#pragma unroll
    for (int j = 0; j < kOut; ++j) part[j] = 0.0f;
    for (int kk = 0; kk < 4; ++kk) {
      const float* r = &pri[(k0 + kk) * kOut];
#pragma unroll
      for (int j = 0; j < kOut; ++j) part[j] += r[j];
    }
#pragma unroll
    for (int j = 0; j < kOut; ++j) {
      float s = waveSum(part[j]);
      if (lane == 0) atomicAdd(&vacc[j], s);
    }
  }
  __syncthreads();
#pragma unroll
  for (int j = 0; j < kOut; ++j) vloc[j] = vacc[j] * (1.0f / kP);
  __syncthreads();

  for (int it = 0; it < kIters; ++it) {
    if (t < kOut) vacc[t] = 0.0f;
    float n2 = 0.0f;
#pragma unroll
    for (int j = 0; j < kOut; ++j) n2 += vloc[j] * vloc[j];
    const float inv = 1.0f / (sqrtf(n2) + 1e-12f);

    float lg[4], lmax = -3.0e38f;
    for (int kk = 0; kk < 4; ++kk) {
      const float* r = &pri[(k0 + kk) * kOut];
      float s = 0.0f;
#pragma unroll
      for (int j = 0; j < kOut; ++j) s += r[j] * vloc[j];
      s *= inv;
      lg[kk] = s;
      lmax = fmaxf(lmax, s);
    }
    lmax = waveMax(lmax);
    if (lane == 0) wred[wv] = lmax;
    __syncthreads();
    float gmax = wred[0];
#pragma unroll
    for (int w = 1; w < 8; ++w) gmax = fmaxf(gmax, wred[w]);
    __syncthreads();

    float den = 0.0f, ws[kOut];
#pragma unroll
    for (int j = 0; j < kOut; ++j) ws[j] = 0.0f;
    for (int kk = 0; kk < 4; ++kk) {
      float pexp = __expf(lg[kk] - gmax);
      den += pexp;
      const float* r = &pri[(k0 + kk) * kOut];
#pragma unroll
      for (int j = 0; j < kOut; ++j) ws[j] += pexp * r[j];
    }
    den = waveSum(den);
    if (lane == 0) wred[wv] = den;
#pragma unroll
    for (int j = 0; j < kOut; ++j) {
      float s = waveSum(ws[j]);
      if (lane == 0) atomicAdd(&vacc[j], s);
    }
    __syncthreads();
    float gden = 0.0f;
#pragma unroll
    for (int w = 0; w < 8; ++w) gden += wred[w];
    const float invd = 1.0f / gden;              // softmax denom: v = sum(exp*pri)/sum(exp)
#pragma unroll
    for (int j = 0; j < kOut; ++j) vloc[j] = vacc[j] * invd;
    __syncthreads();
  }

  if (t == 0) {                                  // ||squash(v)|| written directly
    float n2 = 0.0f;
    for (int j = 0; j < kOut; ++j) n2 += vloc[j] * vloc[j];
    float scale = (n2 / (1.0f + n2)) * rsqrtf(n2 + 1e-12f);
    out[g * kC + o] = scale * sqrtf(n2);
  }
}

// ---------------- host orchestration ----------------

extern "C" void kernel_launch(void* const* d_in, const int* in_sizes, int n_in,
                              void* d_out, int out_size, void* d_ws, size_t ws_size,
                              hipStream_t stream) {
  (void)in_sizes; (void)n_in; (void)out_size; (void)ws_size;
  const float* x     = (const float*)d_in[0];
  const int*   eidx  = (const int*)d_in[1];   // [2, E]
  // d_in[2] = batch (implied by layout, unused)
  const float* Ws[3] = {(const float*)d_in[3], (const float*)d_in[5], (const float*)d_in[7]};
  const float* bs[3] = {(const float*)d_in[4], (const float*)d_in[6], (const float*)d_in[8]};
  const float* capsW = (const float*)d_in[9];
  float* out = (float*)d_out;

  const int* src = eidx;
  const int* dst = eidx + kE;

  // workspace partition (~330 MB)
  float* dis    = (float*)d_ws;                          // kN (deg -> dis in place)
  float* hlin   = dis + kN;                              // kN*kH
  float* Hcat   = hlin + (size_t)kN * kH;                // kN*kIn
  float* pe     = Hcat + (size_t)kN * kIn;               // kP*kIn
  float* priors = pe + (size_t)kP * kIn;                 // kN*kCol
  int*   order  = (int*)(priors + (size_t)kN * kCol);    // kN
  int*   count  = order + kN;                            // kG
  int*   bucket = count + kG;                            // kG*kCap

  // degrees & symmetric normalization
  k_init_deg<<<kN / 256, 256, 0, stream>>>(dis);
  k_edge_deg<<<kE / 256, 256, 0, stream>>>(src, dst, dis);
  k_deg_to_dis<<<kN / 256, 256, 0, stream>>>(dis);

  // bucket edges by graph (once; reused by all 3 layers)
  k_zero_counts<<<1, 256, 0, stream>>>(count);
  k_bucket_fill<<<kE / 256, 256, 0, stream>>>(src, count, bucket);

  // 3 GCN layers: WMMA dense matmul, then per-graph LDS aggregation + tanh
  const float* Xs[3]  = {x, Hcat, Hcat + 32};
  const int    lds[3] = {kFin, kIn, kIn};
  const int    Ks[3]  = {kFin, kH, kH};
  for (int L = 0; L < 3; ++L) {
    k_gemm_h<<<kN / 128, 256, 0, stream>>>(Xs[L], lds[L], Ks[L], Ws[L], hlin);
    k_gcn_agg<<<kG, 256, 0, stream>>>(hlin, dis, src, dst, bucket, count, bs[L],
                                      Hcat, L * kH);
  }

  // sort-pool permutation + position encoding table
  k_posenc<<<(kP * kIn) / 256, 256, 0, stream>>>(pe);
  k_sort<<<kG, 512, 0, stream>>>(Hcat, order);

  // capsule priors GEMM (gather + PE fused into A-fragment load)
  k_gemm_priors<<<kN / 128, 256, 0, stream>>>(Hcat, order, pe, capsW, priors);

  // fused routing + squash + norm
  k_routing<<<kG * kC, 256, 0, stream>>>(priors, out);
}
